// ContrastiveLoss_75299366633823
// MI455X (gfx1250) — matrix operand order repaired and checked by reading, not compiled, not run on hardware
//
#include <hip/hip_runtime.h>
#include <hip/hip_bf16.h>
#include <math.h>
#include <stdint.h>

typedef __attribute__((ext_vector_type(16))) _Float16 v16h;
typedef __attribute__((ext_vector_type(8)))  _Float16 v8h;
typedef __attribute__((ext_vector_type(8)))  float    v8f;

#define NROWS 8192
#define DDIM  512
#define KSTEPS (DDIM / 32)   // 16 WMMA K-steps
#define INV_T 10.0f
#define EPS_F 1e-8f
#define NORM_WAVES 8
#define LOSS_WAVES 2         // 2 waves/block * 32KB A-block = 64KB LDS
#define MTILES 2             // 2 M-tiles (32 rows) per wave
#define AFRAG_V16 512        // v16h elements per M-tile region (16KB / 32B)

// ---------------------------------------------------------------------------
// Kernel 1: L2-normalize each row (fp32 -> f16 matrix in workspace)
// ---------------------------------------------------------------------------
__global__ void cl_normalize_kernel(const float* __restrict__ F,
                                    _Float16* __restrict__ G) {
    const int row = blockIdx.x;
    const float* src = F + (size_t)row * DDIM;
    _Float16*    dst = G + (size_t)row * DDIM;

    const float x0 = src[threadIdx.x];
    const float x1 = src[threadIdx.x + 256];
    float ss = x0 * x0 + x1 * x1;

    #pragma unroll
    for (int off = 16; off > 0; off >>= 1)
        ss += __shfl_xor(ss, off, 32);

    __shared__ float s_part[NORM_WAVES];
    const int wave = threadIdx.x >> 5;
    const int lane = threadIdx.x & 31;
    if (lane == 0) s_part[wave] = ss;
    __syncthreads();

    float tot = 0.0f;
    #pragma unroll
    for (int w = 0; w < NORM_WAVES; ++w) tot += s_part[w];

    const float scale = 1.0f / fmaxf(sqrtf(tot), 1e-12f);
    dst[threadIdx.x]       = (_Float16)(x0 * scale);
    dst[threadIdx.x + 256] = (_Float16)(x1 * scale);
}

// ---------------------------------------------------------------------------
// Kernel 2: fused two-pass contrastive loss, 32x32 register-blocked.
//
// Each wave owns TWO 16-row M-tiles (32 rows). Its A row-block (32x512 f16
// = 32KB) is DMA'd once into private LDS (global_load_async_to_lds_b128),
// pre-permuted to per-fragment layout: frag (mt,kk,lane) at byte offset
// mt*16384 + kk*1024 + lane*32 (a0 = K-lo 16B, a1 = K-hi 16B contiguous).
// Inner loop: per K-step, 2 A fragments (ds) + 2 B fragments (global) feed
// FOUR independent WMMA chains -> ~0.75KB WGP-ingress bytes per WMMA.
//
// A layout (16x32 f16): lanes 0-15 -> M=lane, K={0..15};
//   lanes 16-31 -> M=lane-16, K={16..31}.
// B layout (32x16 f16): lane l -> N=l&15, K={0..15} (l<16) / {16..31}.
// C/D layout: lane l<16 -> (M=r, N=l); lane>=16 -> (M=8+r, N=l-16).
// ---------------------------------------------------------------------------
__global__ __launch_bounds__(32 * LOSS_WAVES)
void cl_loss_kernel(const _Float16* __restrict__ G,
                    const int* __restrict__ labels,
                    float* __restrict__ rowLoss) {
    __shared__ _Float16 aLds[LOSS_WAVES * MTILES * 16 * DDIM];   // 64 KB

    const int wave = threadIdx.x >> 5;
    const int lane = threadIdx.x & 31;
    const int l15  = lane & 15;
    const int half = lane >> 4;                 // 0: rows 0-7, 1: rows 8-15
    const int m0   = (blockIdx.x * LOSS_WAVES + wave) * (MTILES * 16);

    _Float16* aWave = aLds + (size_t)wave * MTILES * 16 * DDIM;

    // ---- async DMA: global A row-block -> LDS, permuted to fragments -----
    #pragma unroll
    for (int mt = 0; mt < MTILES; ++mt) {
        const _Float16* gBase =
            G + (size_t)(m0 + mt * 16 + l15) * DDIM + half * 16;
        _Float16* lBase = aWave + (size_t)mt * 16 * DDIM + lane * 16;
        #pragma unroll
        for (int kk = 0; kk < KSTEPS; ++kk) {
            const uint64_t g0 = (uint64_t)(uintptr_t)(gBase + kk * 32);
            const uint64_t g1 = g0 + 16;                            // K-hi 16B
            const unsigned l0 = (unsigned)(uintptr_t)(lBase + (size_t)kk * 512);
            const unsigned l1 = l0 + 16;
            asm volatile("global_load_async_to_lds_b128 %0, %1, off"
                         :: "v"(l0), "v"(g0) : "memory");
            asm volatile("global_load_async_to_lds_b128 %0, %1, off"
                         :: "v"(l1), "v"(g1) : "memory");
        }
    }
    asm volatile("s_wait_asynccnt 0" ::: "memory");

    // per-lane fragment base; frag (mt,kk) at aV[mt*AFRAG_V16 + kk*32]
    const v16h* aV = (const v16h*)(aWave + lane * 16);

    // ---- pass 1: sum_exp per row -----------------------------------------
    float sumExp[MTILES][8];
    #pragma unroll
    for (int mt = 0; mt < MTILES; ++mt)
        #pragma unroll
        for (int r = 0; r < 8; ++r) sumExp[mt][r] = 0.0f;

    for (int n0 = 0; n0 < NROWS; n0 += 32) {
        const _Float16* b0Base = G + (size_t)(n0 + l15) * DDIM + half * 16;
        const _Float16* b1Base = b0Base + (size_t)16 * DDIM;
        v8f c00 = {}, c01 = {}, c10 = {}, c11 = {};
        #pragma unroll 2
        for (int kk = 0; kk < KSTEPS; ++kk) {
            const v16h a0 = aV[kk * 32];
            const v16h a1 = aV[AFRAG_V16 + kk * 32];
            const v16h b0 = *(const v16h*)(b0Base + kk * 32);
            const v16h b1 = *(const v16h*)(b1Base + kk * 32);
            c00 = __builtin_amdgcn_wmma_f32_16x16x32_f16(
                    false, a0, false, b0, (short)0, c00, false, false);
            c01 = __builtin_amdgcn_wmma_f32_16x16x32_f16(
                    false, a0, false, b1, (short)0, c01, false, false);
            c10 = __builtin_amdgcn_wmma_f32_16x16x32_f16(
                    false, a1, false, b0, (short)0, c10, false, false);
            c11 = __builtin_amdgcn_wmma_f32_16x16x32_f16(
                    false, a1, false, b1, (short)0, c11, false, false);
        }
        const int j0 = n0 + l15;
        const int j1 = j0 + 16;
        #pragma unroll
        for (int r = 0; r < 8; ++r) {
            const int i0 = m0 + half * 8 + r;           // M-tile 0 row
            const int i1 = i0 + 16;                     // M-tile 1 row
            const float e00 = __expf(c00[r] * INV_T);
            const float e01 = __expf(c01[r] * INV_T);
            const float e10 = __expf(c10[r] * INV_T);
            const float e11 = __expf(c11[r] * INV_T);
            sumExp[0][r] += (i0 == j0) ? 0.0f : e00;
            sumExp[0][r] += (i0 == j1) ? 0.0f : e01;
            sumExp[1][r] += (i1 == j0) ? 0.0f : e10;
            sumExp[1][r] += (i1 == j1) ? 0.0f : e11;
        }
    }
    float invSum[MTILES][8];
    #pragma unroll
    for (int mt = 0; mt < MTILES; ++mt)
        #pragma unroll
        for (int r = 0; r < 8; ++r) {
            #pragma unroll
            for (int off = 1; off < 16; off <<= 1)
                sumExp[mt][r] += __shfl_xor(sumExp[mt][r], off, 32);
            invSum[mt][r] = 1.0f / sumExp[mt][r];       // hoisted IEEE divide
        }

    // ---- label-only positive counts (no matmul needed) -------------------
    int labA[MTILES][8];
    #pragma unroll
    for (int mt = 0; mt < MTILES; ++mt)
        #pragma unroll
        for (int r = 0; r < 8; ++r)
            labA[mt][r] = labels[m0 + mt * 16 + half * 8 + r];

    float numPos[MTILES][8];
    {
        float cnt[MTILES][8];
        #pragma unroll
        for (int mt = 0; mt < MTILES; ++mt)
            #pragma unroll
            for (int r = 0; r < 8; ++r) cnt[mt][r] = 0.0f;

        for (int j = l15; j < NROWS; j += 16) {         // each half scans all j
            const int lb = labels[j];
            #pragma unroll
            for (int mt = 0; mt < MTILES; ++mt)
                #pragma unroll
                for (int r = 0; r < 8; ++r)
                    cnt[mt][r] += (labA[mt][r] == lb) ? 1.0f : 0.0f;
        }
        #pragma unroll
        for (int mt = 0; mt < MTILES; ++mt)
            #pragma unroll
            for (int r = 0; r < 8; ++r) {
                #pragma unroll
                for (int off = 1; off < 16; off <<= 1)
                    cnt[mt][r] += __shfl_xor(cnt[mt][r], off, 32);
                numPos[mt][r] = cnt[mt][r] - 1.0f;      // exclude self
            }
    }

    // ---- pass 2: masked loss sums ----------------------------------------
    float lossAcc[MTILES][8];
    #pragma unroll
    for (int mt = 0; mt < MTILES; ++mt)
        #pragma unroll
        for (int r = 0; r < 8; ++r) lossAcc[mt][r] = 0.0f;

    for (int n0 = 0; n0 < NROWS; n0 += 32) {
        const _Float16* b0Base = G + (size_t)(n0 + l15) * DDIM + half * 16;
        const _Float16* b1Base = b0Base + (size_t)16 * DDIM;
        v8f c00 = {}, c01 = {}, c10 = {}, c11 = {};
        #pragma unroll 2
        for (int kk = 0; kk < KSTEPS; ++kk) {
            const v16h a0 = aV[kk * 32];
            const v16h a1 = aV[AFRAG_V16 + kk * 32];
            const v16h b0 = *(const v16h*)(b0Base + kk * 32);
            const v16h b1 = *(const v16h*)(b1Base + kk * 32);
            c00 = __builtin_amdgcn_wmma_f32_16x16x32_f16(
                    false, a0, false, b0, (short)0, c00, false, false);
            c01 = __builtin_amdgcn_wmma_f32_16x16x32_f16(
                    false, a0, false, b1, (short)0, c01, false, false);
            c10 = __builtin_amdgcn_wmma_f32_16x16x32_f16(
                    false, a1, false, b0, (short)0, c10, false, false);
            c11 = __builtin_amdgcn_wmma_f32_16x16x32_f16(
                    false, a1, false, b1, (short)0, c11, false, false);
        }
        const int j0    = n0 + l15;
        const int j1    = j0 + 16;
        const int labB0 = labels[j0];
        const int labB1 = labels[j1];
        #pragma unroll
        for (int r = 0; r < 8; ++r) {
            const int i0 = m0 + half * 8 + r;
            const int i1 = i0 + 16;
            const float p00 = ((labA[0][r] == labB0) && (i0 != j0)) ? 1.0f : 0.0f;
            const float p01 = ((labA[0][r] == labB1) && (i0 != j1)) ? 1.0f : 0.0f;
            const float p10 = ((labA[1][r] == labB0) && (i1 != j0)) ? 1.0f : 0.0f;
            const float p11 = ((labA[1][r] == labB1) && (i1 != j1)) ? 1.0f : 0.0f;
            const float q00 = __expf(c00[r] * INV_T) * invSum[0][r];
            const float q01 = __expf(c01[r] * INV_T) * invSum[0][r];
            const float q10 = __expf(c10[r] * INV_T) * invSum[1][r];
            const float q11 = __expf(c11[r] * INV_T) * invSum[1][r];
            lossAcc[0][r] += p00 * (-__logf(q00 + EPS_F))
                           + p01 * (-__logf(q01 + EPS_F));
            lossAcc[1][r] += p10 * (-__logf(q10 + EPS_F))
                           + p11 * (-__logf(q11 + EPS_F));
        }
    }
    #pragma unroll
    for (int mt = 0; mt < MTILES; ++mt)
        #pragma unroll
        for (int r = 0; r < 8; ++r)
            #pragma unroll
            for (int off = 1; off < 16; off <<= 1)
                lossAcc[mt][r] += __shfl_xor(lossAcc[mt][r], off, 32);

    if (l15 == 0) {
        #pragma unroll
        for (int mt = 0; mt < MTILES; ++mt)
            #pragma unroll
            for (int r = 0; r < 8; ++r)
                rowLoss[m0 + mt * 16 + half * 8 + r] =
                    lossAcc[mt][r] / (numPos[mt][r] + EPS_F);
    }
}

// ---------------------------------------------------------------------------
// Kernel 3: mean of per-row losses -> d_out[0]
// ---------------------------------------------------------------------------
__global__ void cl_mean_kernel(const float* __restrict__ rowLoss,
                               float* __restrict__ out) {
    float acc = 0.0f;
    for (int i = threadIdx.x; i < NROWS; i += 256) acc += rowLoss[i];

    #pragma unroll
    for (int off = 16; off > 0; off >>= 1)
        acc += __shfl_xor(acc, off, 32);

    __shared__ float s_part[NORM_WAVES];
    const int wave = threadIdx.x >> 5;
    const int lane = threadIdx.x & 31;
    if (lane == 0) s_part[wave] = acc;
    __syncthreads();

    if (threadIdx.x == 0) {
        float tot = 0.0f;
        #pragma unroll
        for (int w = 0; w < NORM_WAVES; ++w) tot += s_part[w];
        out[0] = tot / (float)NROWS;
    }
}

// ---------------------------------------------------------------------------
extern "C" void kernel_launch(void* const* d_in, const int* in_sizes, int n_in,
                              void* d_out, int out_size, void* d_ws, size_t ws_size,
                              hipStream_t stream) {
    (void)in_sizes; (void)n_in; (void)out_size; (void)ws_size;

    const float* features = (const float*)d_in[0];
    const int*   labels   = (const int*)d_in[1];
    float*       out      = (float*)d_out;

    char* ws = (char*)d_ws;
    _Float16* G       = (_Float16*)ws;                                   // 8 MB
    float*    rowLoss = (float*)(ws + (size_t)NROWS * DDIM * sizeof(_Float16));

    // 1) normalize rows fp32 -> f16 matrix
    cl_normalize_kernel<<<NROWS, 256, 0, stream>>>(features, G);

    // 2) fused two-pass WMMA contrastive loss, 32x32 register-blocked
    cl_loss_kernel<<<NROWS / (MTILES * 16) / LOSS_WAVES, 32 * LOSS_WAVES, 0,
                     stream>>>(G, labels, rowLoss);

    // 3) mean reduce
    cl_mean_kernel<<<1, 256, 0, stream>>>(rowLoss, out);
}
